// ZXRepairNet_65317862637649
// MI455X (gfx1250) — compile-verified, matrix operand order checked by reading, not run on hardware
//
#include <hip/hip_runtime.h>
#include <hip/hip_fp16.h>

typedef __attribute__((ext_vector_type(16))) _Float16 v16h;
typedef __attribute__((ext_vector_type(8)))  _Float16 v8h;
typedef __attribute__((ext_vector_type(8)))  float    v8f;

#define BN_EPS 1e-5f

// ---------------- degree / normalization ----------------
__global__ void deg_kernel(const float* __restrict__ adj, float* __restrict__ dinv) {
  __shared__ float red[256];
  int row = blockIdx.x;                       // b*256 + i
  red[threadIdx.x] = adj[(size_t)row * 256 + threadIdx.x];
  __syncthreads();
  for (int off = 128; off > 0; off >>= 1) {
    if (threadIdx.x < off) red[threadIdx.x] += red[threadIdx.x + off];
    __syncthreads();
  }
  if (threadIdx.x == 0) {
    float d = red[0] + 1.0f;                  // +1 self loop
    d = fmaxf(d, 1.0f);                       // clip(.,1)
    dinv[row] = rsqrtf(d);
  }
}

// ---------------- generic dense linear: out = opt(in@W [+bias]) [*dinv] [relu] ----
__global__ void linear_kernel(const float* __restrict__ in, const float* __restrict__ W,
                              const float* __restrict__ bias, const float* __restrict__ dinv,
                              float* __restrict__ out, int rows, int Cin, int Cout, int relu) {
  int idx = blockIdx.x * blockDim.x + threadIdx.x;
  if (idx >= rows * Cout) return;
  int r = idx / Cout, c = idx - r * Cout;
  const float* ip = in + (size_t)r * Cin;
  float acc = bias ? bias[c] : 0.f;
  for (int k = 0; k < Cin; k++) acc = fmaf(ip[k], W[(size_t)k * Cout + c], acc);
  if (dinv) acc *= dinv[r];
  if (relu) acc = fmaxf(acc, 0.f);
  out[idx] = acc;
}

// ---- linear producing f16 output transposed per batch: xsT[(b*Cout+c)*256 + n] ----
// xs = dinv[b,n] * (in @ W)   (feeds the WMMA adjacency GEMM's B operand)
__global__ void linear_t_f16_kernel(const float* __restrict__ in, const float* __restrict__ W,
                                    const float* __restrict__ dinv, _Float16* __restrict__ xsT,
                                    int rows, int Cin, int Cout) {
  int idx = blockIdx.x * blockDim.x + threadIdx.x;
  if (idx >= rows * Cout) return;
  int r = idx / Cout, c = idx - r * Cout;
  const float* ip = in + (size_t)r * Cin;
  float acc = 0.f;
  for (int k = 0; k < Cin; k++) acc = fmaf(ip[k], W[(size_t)k * Cout + c], acc);
  acc *= dinv[r];
  int b = r >> 8, n = r & 255;
  xsT[((size_t)b * Cout + c) * 256 + n] = (_Float16)acc;
}

// ---------------- WMMA batched adjacency GEMM (C compile-time; xs f16 transposed) ----
// out[b,i,c] = relu?( dinv[b,i]*( sum_j adj[b,i,j]*xs[b,j,c] + xs[b,i,c] ) + bias[c] )
template <int C>
__global__ void adjgemm_wmma(const float* __restrict__ adj, const _Float16* __restrict__ xsT,
                             const float* __restrict__ dinv, const float* __restrict__ bias,
                             float* __restrict__ out, int relu) {
  __shared__ __align__(16) _Float16 sAdj[16 * 256];   // 8 KB (adj values 0/1/2 exact in f16)
  int bb = blockIdx.x >> 4;                  // batch
  int i0 = (blockIdx.x & 15) * 16;           // i tile base
  const float* arow = adj + ((size_t)bb * 256 + i0) * 256;
  __builtin_prefetch(arow, 0, 3);
  for (int t = threadIdx.x; t < 16 * 256; t += 256) sAdj[t] = (_Float16)arow[t];
  __syncthreads();

  int lane = threadIdx.x & 31, w = threadIdx.x >> 5;
  constexpr int ntiles = C >> 4;
  int M = lane & 15;
  int lhalf = (lane & 16) ? 8 : 0;
  for (int nt = w; nt < ntiles; nt += 8) {
    int c = nt * 16 + (lane & 15);
    const _Float16* xcol = xsT + ((size_t)bb * C + c) * 256;   // column c, contiguous in n
    v8f acc = {};
#pragma unroll 2
    for (int kc = 0; kc < 8; kc++) {         // K = 256
      // A: two contiguous 8xf16 runs from the adj tile -> ds_load_b128 x2
      const _Float16* ap = sAdj + M * 256 + kc * 32 + lhalf;
      v16h A = __builtin_shufflevector(*(const v8h*)ap, *(const v8h*)(ap + 16),
                                       0, 1, 2, 3, 4, 5, 6, 7,
                                       8, 9, 10, 11, 12, 13, 14, 15);
      // B: two contiguous 8xf16 runs from transposed xs -> global_load_b128 x2
      const _Float16* bp = xcol + kc * 32 + lhalf;
      v16h Bf = __builtin_shufflevector(*(const v8h*)bp, *(const v8h*)(bp + 16),
                                        0, 1, 2, 3, 4, 5, 6, 7,
                                        8, 9, 10, 11, 12, 13, 14, 15);
      acc = __builtin_amdgcn_wmma_f32_16x16x32_f16(false, A, false, Bf,
                                                   (short)0, acc, false, false);
    }
    int rbase = (lane & 16) ? 8 : 0;
    float* op = out + (size_t)(bb * 256 + i0 + rbase) * C + c;
    float bval = bias[c];
#pragma unroll
    for (int v = 0; v < 8; v++) {
      float self = (float)xcol[i0 + rbase + v];               // +I self-loop term
      float val = dinv[bb * 256 + i0 + rbase + v] * (acc[v] + self) + bval;
      if (relu) val = fmaxf(val, 0.f);
      op[v * C] = val;
    }
  }
}

// ---------------- BatchNorm (training-mode, biased var) ----------------
__global__ void bnstats_kernel(const float* __restrict__ t, float* __restrict__ mean,
                               float* __restrict__ var, int rows, int C) {
  __shared__ float s1[256], s2[256];
  int c = blockIdx.x;
  float a = 0.f, b = 0.f;
  for (int r = threadIdx.x; r < rows; r += 256) {
    float v = t[(size_t)r * C + c];
    a += v; b += v * v;
  }
  s1[threadIdx.x] = a; s2[threadIdx.x] = b;
  __syncthreads();
  for (int off = 128; off > 0; off >>= 1) {
    if (threadIdx.x < off) { s1[threadIdx.x] += s1[threadIdx.x + off];
                             s2[threadIdx.x] += s2[threadIdx.x + off]; }
    __syncthreads();
  }
  if (threadIdx.x == 0) {
    float m = s1[0] / rows;
    mean[c] = m;
    var[c]  = s2[0] / rows - m * m;
  }
}

__global__ void bnapply_kernel(const float* __restrict__ t, const float* __restrict__ mean,
                               const float* __restrict__ var, const float* __restrict__ g,
                               const float* __restrict__ beta, float* __restrict__ h,
                               int total, int C) {
  int idx = blockIdx.x * 256 + threadIdx.x;
  if (idx >= total) return;
  int c = idx % C;
  float v = (t[idx] - mean[c]) * rsqrtf(var[c] + BN_EPS) * g[c] + beta[c];
  h[idx] = fmaxf(v, 0.f);
}

// ---------------- pairwise adjacency decoder (dominant stage) ----------------
// block = (i-tile, j-tile, b); 256 pairs; p=relu(hi+hj), q=relu(p@e2w+e2b), logit=q@e3w+e3b
__global__ void decoder_kernel(const float* __restrict__ hi, const float* __restrict__ hj,
                               const float* __restrict__ e2w, const float* __restrict__ e2b,
                               const float* __restrict__ e3w, const float* __restrict__ e3b,
                               float* __restrict__ logits) {
  __shared__ __align__(16) _Float16 sHi[16 * 128];   // 4 KB (e1b folded into hi)
  __shared__ __align__(16) _Float16 sHj[16 * 128];   // 4 KB
  __shared__ __align__(16) _Float16 sE2t[64 * 128];  // 16 KB (e2w f16, transposed [c][k])
  __shared__ __align__(16) _Float16 sQ[256 * 64];    // 32 KB
  int it = blockIdx.x, jt = blockIdx.y, bb = blockIdx.z;
  const float* hp = hi + ((size_t)bb * 256 + it * 16) * 128;
  const float* jp = hj + ((size_t)bb * 256 + jt * 16) * 128;
  __builtin_prefetch(hp, 0, 3);
  __builtin_prefetch(jp, 0, 3);
  __builtin_prefetch(e2w, 0, 3);
  for (int t = threadIdx.x; t < 16 * 128; t += 256) {
    sHi[t] = (_Float16)hp[t];
    sHj[t] = (_Float16)jp[t];
  }
  // coalesced read of e2w[k][c], transposed write to sE2t[c][k]
  for (int t = threadIdx.x; t < 128 * 64; t += 256) {
    int k = t >> 6, c = t & 63;
    sE2t[c * 128 + k] = (_Float16)e2w[t];
  }
  __syncthreads();

  int lane = threadIdx.x & 31, w = threadIdx.x >> 5;
  int lhalf = (lane & 16) ? 8 : 0;

  // Preload all 16 (kc,nt) B fragments: 2x ds_load_b128 each from transposed e2w.
  v16h Bfr[16];
#pragma unroll
  for (int kc = 0; kc < 4; kc++) {
#pragma unroll
    for (int nt = 0; nt < 4; nt++) {
      int c = nt * 16 + (lane & 15);
      const _Float16* bp = sE2t + c * 128 + kc * 32 + lhalf;
      Bfr[kc * 4 + nt] = __builtin_shufflevector(*(const v8h*)bp, *(const v8h*)(bp + 16),
                                                 0, 1, 2, 3, 4, 5, 6, 7,
                                                 8, 9, 10, 11, 12, 13, 14, 15);
    }
  }

  int jj = lane & 15;
  const v8h hz = {};
  for (int mi = 0; mi < 2; mi++) {
    int m = w + mi * 8;                      // ii row owned by this wave
    v8f acc[4] = {};
#pragma unroll
    for (int kc = 0; kc < 4; kc++) {         // K = 128
      // two contiguous v8h runs from each of sHi (row m) and sHj (row jj)
      const _Float16* hix = sHi + m * 128 + kc * 32 + lhalf;
      const _Float16* hjx = sHj + jj * 128 + kc * 32 + lhalf;
      v8h s0 = *(const v8h*)hix + *(const v8h*)hjx;               // p = hi+hj+e1b
      v8h s1 = *(const v8h*)(hix + 16) + *(const v8h*)(hjx + 16);
      s0 = __builtin_elementwise_max(s0, hz);                     // relu -> v_pk_max_num_f16
      s1 = __builtin_elementwise_max(s1, hz);
      v16h A = __builtin_shufflevector(s0, s1, 0, 1, 2, 3, 4, 5, 6, 7,
                                       8, 9, 10, 11, 12, 13, 14, 15);
#pragma unroll
      for (int nt = 0; nt < 4; nt++) {       // N = 64
        acc[nt] = __builtin_amdgcn_wmma_f32_16x16x32_f16(false, A, false, Bfr[kc * 4 + nt],
                                                         (short)0, acc[nt], false, false);
      }
    }
    int rbase = (lane & 16) ? 8 : 0;
#pragma unroll
    for (int nt = 0; nt < 4; nt++) {
      int c = nt * 16 + (lane & 15);
      float bval = e2b[c];
#pragma unroll
      for (int v = 0; v < 8; v++) {
        sQ[(m * 16 + rbase + v) * 64 + c] = (_Float16)fmaxf(acc[nt][v] + bval, 0.f);
      }
    }
  }
  __syncthreads();
  // phase B: one thread per pair, 64-wide dot with e3w (8x ds_load_b128)
  int P = threadIdx.x;
  const v8h* qp = (const v8h*)(sQ + P * 64);
  float s = e3b[0];
#pragma unroll
  for (int t = 0; t < 8; t++) {
    v8h q = qp[t];
#pragma unroll
    for (int e = 0; e < 8; e++) s += (float)q[e] * e3w[t * 8 + e];
  }
  int ii = P >> 4;
  logits[((size_t)bb * 256 + it * 16 + ii) * 256 + jt * 16 + (P & 15)] = s;
}

// ---------------- symmetrize + sigmoid ----------------
__global__ void sigsym_kernel(const float* __restrict__ L, float* __restrict__ out) {
  int idx = blockIdx.x * 256 + threadIdx.x;
  if (idx >= 4 * 256 * 256) return;
  int bb = idx >> 16, ij = idx & 65535, i = ij >> 8, j = ij & 255;
  float x = 0.5f * (L[idx] + L[(bb << 16) + (j << 8) + i]);
  out[idx] = 1.f / (1.f + expf(-x));
}

extern "C" void kernel_launch(void* const* d_in, const int* in_sizes, int n_in,
                              void* d_out, int out_size, void* d_ws, size_t ws_size,
                              hipStream_t stream) {
  const float* x   = (const float*)d_in[0];
  const float* adj = (const float*)d_in[1];
  const float* W1  = (const float*)d_in[2];
  const float* b1  = (const float*)d_in[3];
  const float* W2  = (const float*)d_in[4];
  const float* b2  = (const float*)d_in[5];
  const float* W3  = (const float*)d_in[6];
  const float* b3  = (const float*)d_in[7];
  const float* g1  = (const float*)d_in[8];
  const float* be1 = (const float*)d_in[9];
  const float* g2  = (const float*)d_in[10];
  const float* be2 = (const float*)d_in[11];
  const float* e1w = (const float*)d_in[12];
  const float* e1b = (const float*)d_in[13];
  const float* e2w = (const float*)d_in[14];
  const float* e2b = (const float*)d_in[15];
  const float* e3w = (const float*)d_in[16];
  const float* e3b = (const float*)d_in[17];
  const float* f1w = (const float*)d_in[18];
  const float* f1b = (const float*)d_in[19];
  const float* f2w = (const float*)d_in[20];
  const float* f2b = (const float*)d_in[21];
  const float* f3w = (const float*)d_in[22];
  const float* f3b = (const float*)d_in[23];

  float* ws      = (float*)d_ws;
  float* dinv    = ws;                 // 1024
  float* tbuf    = dinv + 1024;        // 262144 (pre-BN / h3)
  float* hbuf    = tbuf + 262144;      // 262144
  float* xsbuf   = hbuf + 262144;      // 262144 floats (used as f16 transposed xs)
  float* hibuf   = xsbuf + 262144;     // 131072
  float* hjbuf   = hibuf + 131072;     // 131072
  float* meanb   = hjbuf + 131072;     // 128
  float* varb    = meanb + 128;        // 128
  float* logitsb = varb + 128;         // 262144
  float* ft1     = logitsb + 262144;   // 131072
  float* ft2     = ft1 + 131072;       // 131072
  _Float16* xsT  = (_Float16*)xsbuf;   // [b][C][256] f16

  const int rows = 4 * 256;

  deg_kernel<<<1024, 256, 0, stream>>>(adj, dinv);

  // layer 1: xsT = dinv*(x@W1) (f16,T); t = adjGEMM + b1; BN + relu
  linear_t_f16_kernel<<<(rows * 128 + 255) / 256, 256, 0, stream>>>(x, W1, dinv, xsT, rows, 6, 128);
  adjgemm_wmma<128><<<64, 256, 0, stream>>>(adj, xsT, dinv, b1, tbuf, 0);
  bnstats_kernel<<<128, 256, 0, stream>>>(tbuf, meanb, varb, rows, 128);
  bnapply_kernel<<<(rows * 128 + 255) / 256, 256, 0, stream>>>(tbuf, meanb, varb, g1, be1, hbuf, rows * 128, 128);

  // layer 2
  linear_t_f16_kernel<<<(rows * 128 + 255) / 256, 256, 0, stream>>>(hbuf, W2, dinv, xsT, rows, 128, 128);
  adjgemm_wmma<128><<<64, 256, 0, stream>>>(adj, xsT, dinv, b2, tbuf, 0);
  bnstats_kernel<<<128, 256, 0, stream>>>(tbuf, meanb, varb, rows, 128);
  bnapply_kernel<<<(rows * 128 + 255) / 256, 256, 0, stream>>>(tbuf, meanb, varb, g2, be2, hbuf, rows * 128, 128);

  // layer 3 (relu only) -> h3 in tbuf
  linear_t_f16_kernel<<<(rows * 256 + 255) / 256, 256, 0, stream>>>(hbuf, W3, dinv, xsT, rows, 128, 256);
  adjgemm_wmma<256><<<64, 256, 0, stream>>>(adj, xsT, dinv, b3, tbuf, 1);

  // adjacency decoder: hi = h3@e1w[:256] + e1b ; hj = h3@e1w[256:]
  linear_kernel<<<(rows * 128 + 255) / 256, 256, 0, stream>>>(tbuf, e1w, e1b, nullptr, hibuf, rows, 256, 128, 0);
  linear_kernel<<<(rows * 128 + 255) / 256, 256, 0, stream>>>(tbuf, e1w + 256 * 128, nullptr, nullptr, hjbuf, rows, 256, 128, 0);
  dim3 dg(16, 16, 4);
  decoder_kernel<<<dg, 256, 0, stream>>>(hibuf, hjbuf, e2w, e2b, e3w, e3b, logitsb);
  sigsym_kernel<<<(262144 + 255) / 256, 256, 0, stream>>>(logitsb, (float*)d_out);

  // feature decoder
  linear_kernel<<<(rows * 128 + 255) / 256, 256, 0, stream>>>(tbuf, f1w, f1b, nullptr, ft1, rows, 256, 128, 1);
  linear_kernel<<<(rows * 128 + 255) / 256, 256, 0, stream>>>(ft1, f2w, f2b, nullptr, ft2, rows, 128, 128, 1);
  linear_kernel<<<(rows * 6 + 255) / 256, 256, 0, stream>>>(ft2, f3w, f3b, nullptr, (float*)d_out + 262144, rows, 128, 6, 0);
}
